// LLamaAttentionBlock_29592324669790
// MI455X (gfx1250) — matrix operand order repaired and checked by reading, not compiled, not run on hardware
//
#include <hip/hip_runtime.h>
#include <hip/hip_bf16.h>

typedef __attribute__((ext_vector_type(16))) _Float16 v16h;
typedef __attribute__((ext_vector_type(8)))  _Float16 v8h;
typedef __attribute__((ext_vector_type(8)))  float    v8f;
typedef __attribute__((ext_vector_type(4)))  int      v4i;

typedef __attribute__((address_space(1))) v4i* as1_v4i_p;
typedef __attribute__((address_space(3))) v4i* as3_v4i_p;

#define H_ 32
#define D_ 128
#define NEG_INF (-1e9f)
#define SCALE 0.08838834764831845f   // 1/sqrt(128)

#define KT_PITCH 136   // 128 + 8 halves: fragment reads hit distinct LDS banks
#define VT_PITCH 72    // 64 + 8 halves

// ---- CDNA5 async global->LDS staging (guarded; sync fallback) --------------
__device__ __forceinline__ void async_copy16(const _Float16* g, _Float16* l) {
#if __has_builtin(__builtin_amdgcn_global_load_async_to_lds_b128)
    __builtin_amdgcn_global_load_async_to_lds_b128(
        (as1_v4i_p)g, (as3_v4i_p)l, 0, 0);
#else
    *(v8h*)l = *(const v8h*)g;
#endif
}

__device__ __forceinline__ void wait_async0() {
#if __has_builtin(__builtin_amdgcn_s_wait_asynccnt)
    __builtin_amdgcn_s_wait_asynccnt(0);
#else
    asm volatile("s_wait_asynccnt 0" ::: "memory");
#endif
}

// ---------------------------------------------------------------------------
// Kernels 1/2: inverse slot map in workspace: inv[slot] = token or -1
// ---------------------------------------------------------------------------
__global__ void init_inv_kernel(int* __restrict__ inv, int n) {
    int i = blockIdx.x * blockDim.x + threadIdx.x;
    if (i < n) inv[i] = -1;
}

__global__ void scatter_inv_kernel(const int* __restrict__ save_slots,
                                   int* __restrict__ inv, int T) {
    int i = blockIdx.x * blockDim.x + threadIdx.x;
    if (i < T) inv[save_slots[i]] = i;
}

// ---------------------------------------------------------------------------
// Kernel 3: gather + RoPE + f16 convert + V transpose (bandwidth-bound).
//   Kf: [B][H][S][D] f16 (rope'd keys)       -> K B-fragments contiguous
//   Vt: [B][H][D][S] f16 (values transposed) -> V B-fragments contiguous
// ---------------------------------------------------------------------------
__global__ __launch_bounds__(128)
void gather_kv_kernel(const float* __restrict__ K,  const float* __restrict__ V,
                      const float* __restrict__ Kc, const float* __restrict__ Vc,
                      const float* __restrict__ cosT, const float* __restrict__ sinT,
                      const int* __restrict__ fetch, const int* __restrict__ emb,
                      const int* __restrict__ inv,
                      _Float16* __restrict__ Kf, _Float16* __restrict__ Vt, int S)
{
    const int h  = blockIdx.y, b = blockIdx.z;
    const int jt = threadIdx.x >> 2;            // token within 32-token tile
    const int qd = threadIdx.x & 3;             // feature quarter (32 feats)
    const int jpos = blockIdx.x * 32 + jt;
    const int s  = fetch[b * S + jpos];
    const int t  = inv[s];

    __shared__ _Float16 vtile[D_][34];          // padded against bank conflicts

    const int f0  = qd * 32;
    const int pf0 = (f0 < 64) ? f0 + 64 : f0 - 64;   // rotate_half partner
    const float sg = (f0 < 64) ? -1.f : 1.f;

    float kv[32], vv[32];
    if (t >= 0) {                                // fresh token: apply RoPE
        const float* kp = K + (size_t)t * H_ * D_ + h * D_;
        const float* vp = V + (size_t)t * H_ * D_ + h * D_;
        const int    p  = emb[t];
        const float* ck = cosT + (size_t)p * D_;
        const float* sk = sinT + (size_t)p * D_;
        #pragma unroll
        for (int e = 0; e < 32; ++e) {
            kv[e] = kp[f0 + e] * ck[f0 + e] + sg * kp[pf0 + e] * sk[f0 + e];
            vv[e] = vp[f0 + e];
        }
    } else {                                     // stale slot: cached rows
        const float* kp = Kc + (size_t)s * H_ * D_ + h * D_;
        const float* vp = Vc + (size_t)s * H_ * D_ + h * D_;
        #pragma unroll
        for (int e = 0; e < 32; ++e) { kv[e] = kp[f0 + e]; vv[e] = vp[f0 + e]; }
    }

    _Float16* ko = Kf + ((size_t)(b * H_ + h) * S + jpos) * D_ + f0;
    #pragma unroll
    for (int e = 0; e < 32; ++e) ko[e] = (_Float16)kv[e];

    #pragma unroll
    for (int e = 0; e < 32; ++e) vtile[f0 + e][jt] = (_Float16)vv[e];
    __syncthreads();

    {   // coalesced transposed write-out: thread d owns one feature row
        const int d = threadIdx.x;
        _Float16* vo = Vt + ((size_t)(b * H_ + h) * D_ + d) * S + blockIdx.x * 32;
        #pragma unroll
        for (int e = 0; e < 32; ++e) vo[e] = vtile[d][e];
    }
}

// ---------------------------------------------------------------------------
// Kernel 4: flash-attention prefill main loop.
//   grid = (S/64, H, B), block = 128 (4 waves), wave owns a 16-row Q tile.
//   Block-uniform 64-key chunks: K/V tiles async-staged into double-buffered
//   LDS once per block and shared by all 4 waves (4x L2 traffic reduction).
//   Per chunk per wave: 16 QK WMMAs + 18 PV/ones WMMAs, one softmax pass.
//   Only the final (diagonal) chunk applies the causal mask.
// ---------------------------------------------------------------------------
__global__ __launch_bounds__(128)
void attn_kernel(const float* __restrict__ Q, const float* __restrict__ cosT,
                 const float* __restrict__ sinT, const int* __restrict__ emb,
                 const _Float16* __restrict__ Kf, const _Float16* __restrict__ Vt,
                 float* __restrict__ out, int S)
{
    const int w   = threadIdx.x >> 5;
    const int lid = threadIdx.x & 31;
    const int hi  = lid >> 4;
    const int ln  = lid & 15;
    const int h   = blockIdx.y;
    const int b   = blockIdx.z;
    const int blockQ0 = blockIdx.x * 64;
    const int q0  = blockQ0 + w * 16;

    __shared__ __align__(16) _Float16 Kt[2][64][KT_PITCH];    // staged K tiles
    __shared__ __align__(16) _Float16 Vl[2][D_][VT_PITCH];    // staged V tiles
    __shared__ __align__(16) _Float16 pbuf[4][16][64];        // wave P tiles

    const _Float16* kpanel = Kf + (size_t)(b * H_ + h) * S * D_;
    const _Float16* vpanel = Vt + (size_t)(b * H_ + h) * D_ * S;

    // ---- Q A-fragments with RoPE; softmax scale folded in -----------------
    const int tq = b * S + q0 + ln;
    const int pq = emb[tq];
    const float* qp = Q    + (size_t)tq * H_ * D_ + h * D_;
    const float* cq = cosT + (size_t)pq * D_;
    const float* sq = sinT + (size_t)pq * D_;

    v16h a[4];
    #pragma unroll
    for (int c = 0; c < 4; ++c) {
        #pragma unroll
        for (int half = 0; half < 2; ++half) {
            const int d0 = c * 32 + hi * 16 + half * 8;
            const int pd = (d0 < 64) ? d0 + 64 : d0 - 64;
            const float sg = (d0 < 64) ? -1.f : 1.f;
            #pragma unroll
            for (int e = 0; e < 8; ++e) {
                float r = qp[d0 + e] * cq[d0 + e] + sg * qp[pd + e] * sq[d0 + e];
                a[c][half * 8 + e] = (_Float16)(r * SCALE);
            }
        }
    }

    float m[8];
    v8f acc[8], accL;                    // accL: row sums via ones-column WMMA
    #pragma unroll
    for (int i = 0; i < 8; ++i) m[i] = -3.0e38f;
    { v8f z = {}; accL = z;
      #pragma unroll
      for (int n = 0; n < 8; ++n) acc[n] = z; }

    v16h vones;
    #pragma unroll
    for (int e = 0; e < 16; ++e) vones[e] = (_Float16)1.0f;

    // ---- cooperative async staging of one 64-key K/V tile pair ------------
    auto stage = [&](int c) {
        const int buf = c & 1;
        const int j0  = c * 64;
        const int tid = threadIdx.x;
        #pragma unroll
        for (int i = 0; i < 8; ++i) {            // K: 64 rows x 128 halves
            const int ch  = tid + i * 128;       // 1024 x 16B chunks
            const int row = ch >> 4, cc = ch & 15;
            async_copy16(kpanel + (size_t)(j0 + row) * D_ + cc * 8,
                         &Kt[buf][row][cc * 8]);
        }
        #pragma unroll
        for (int i = 0; i < 8; ++i) {            // V: 128 rows x 64 halves
            const int ch  = tid + i * 128;
            const int row = ch >> 3, cc = ch & 7;
            async_copy16(vpanel + (size_t)row * S + j0 + cc * 8,
                         &Vl[buf][row][cc * 8]);
        }
    };

    const int nChunks = blockIdx.x + 1;
    stage(0);

    for (int c = 0; c < nChunks; ++c) {
        wait_async0();
        __syncthreads();                         // tile c visible to all waves
        if (c + 1 < nChunks) stage(c + 1);       // prefetch into other buffer
        const int  buf    = c & 1;
        const int  j0     = c * 64;
        const bool masked = (c == nChunks - 1);  // block-diagonal chunk

        // -- QK^T: 4 sub-tiles x 4 chained K=32 WMMAs, K frags from LDS -----
        v8f sc[4];
        #pragma unroll
        for (int kk = 0; kk < 4; ++kk) {
            const _Float16* kp = &Kt[buf][kk * 16 + ln][0];
            v16h bk[4];
            #pragma unroll
            for (int cc = 0; cc < 4; ++cc)
                bk[cc] = *(const v16h*)(kp + cc * 32 + hi * 16);
            v8f cacc = {};
            #pragma unroll
            for (int cc = 0; cc < 4; ++cc)
                cacc = __builtin_amdgcn_wmma_f32_16x16x32_f16(
                    false, a[cc], false, bk[cc], (short)0, cacc, false, false);
            sc[kk] = cacc;
        }

        if (masked) {
            #pragma unroll
            for (int kk = 0; kk < 4; ++kk) {
                const int col = j0 + kk * 16 + ln;
                #pragma unroll
                for (int i = 0; i < 8; ++i) {
                    const int row = q0 + i + hi * 8;
                    if (col > row) sc[kk][i] = NEG_INF;
                }
            }
        }

        // -- online softmax: batched butterfly max (step-major) -------------
        float rm[8];
        #pragma unroll
        for (int i = 0; i < 8; ++i)
            rm[i] = fmaxf(fmaxf(sc[0][i], sc[1][i]), fmaxf(sc[2][i], sc[3][i]));
        #pragma unroll
        for (int ofs = 1; ofs < 16; ofs <<= 1) {
            float tmp[8];
            #pragma unroll
            for (int i = 0; i < 8; ++i) tmp[i] = __shfl_xor(rm[i], ofs, 32);
            #pragma unroll
            for (int i = 0; i < 8; ++i) rm[i] = fmaxf(rm[i], tmp[i]);
        }
        float corr[8];
        #pragma unroll
        for (int i = 0; i < 8; ++i) {
            const float mnew = fmaxf(m[i], rm[i]);
            corr[i] = __expf(m[i] - mnew);
            m[i] = mnew;
            #pragma unroll
            for (int kk = 0; kk < 4; ++kk)
                sc[kk][i] = __expf(sc[kk][i] - mnew);
        }
        #pragma unroll
        for (int i = 0; i < 8; ++i) accL[i] *= corr[i];
        #pragma unroll
        for (int n = 0; n < 8; ++n)
            #pragma unroll
            for (int i = 0; i < 8; ++i)
                acc[n][i] *= corr[i];

        // -- P: C-layout f32 -> A-layout f16 via wave-private LDS tile ------
        #pragma unroll
        for (int kk = 0; kk < 4; ++kk)
            #pragma unroll
            for (int i = 0; i < 8; ++i)
                pbuf[w][i + hi * 8][kk * 16 + ln] = (_Float16)sc[kk][i];
        asm volatile("s_wait_dscnt 0" ::: "memory");

        // -- P.V (+ ones-column row sums): V frags from LDS ------------------
        #pragma unroll
        for (int t = 0; t < 2; ++t) {
            v16h pa;
            {
                v8h lo = *(const v8h*)&pbuf[w][ln][t * 32 + hi * 16];
                v8h hv = *(const v8h*)&pbuf[w][ln][t * 32 + hi * 16 + 8];
                #pragma unroll
                for (int e = 0; e < 8; ++e) { pa[e] = lo[e]; pa[8 + e] = hv[e]; }
            }
            #pragma unroll
            for (int n = 0; n < 8; ++n) {
                v16h bv = *(const v16h*)&Vl[buf][n * 16 + ln][t * 32 + hi * 16];
                acc[n] = __builtin_amdgcn_wmma_f32_16x16x32_f16(
                    false, pa, false, bv, (short)0, acc[n], false, false);
            }
            accL = __builtin_amdgcn_wmma_f32_16x16x32_f16(
                false, pa, false, vones, (short)0, accL, false, false);
        }
    }

    // ---- epilogue ----------------------------------------------------------
    #pragma unroll
    for (int i = 0; i < 8; ++i) {
        const int row = q0 + i + hi * 8;
        const float invl = 1.0f / accL[i];
        float* op = out + (size_t)(b * S + row) * H_ * D_ + h * D_;
        #pragma unroll
        for (int n = 0; n < 8; ++n)
            op[n * 16 + ln] = acc[n][i] * invl;
    }
}

// ---------------------------------------------------------------------------
extern "C" void kernel_launch(void* const* d_in, const int* in_sizes, int n_in,
                              void* d_out, int out_size, void* d_ws, size_t ws_size,
                              hipStream_t stream) {
    const float* Q      = (const float*)d_in[0];
    const float* K      = (const float*)d_in[1];
    const float* V      = (const float*)d_in[2];
    const float* Kcache = (const float*)d_in[3];
    const float* Vcache = (const float*)d_in[4];
    const float* cosT   = (const float*)d_in[5];
    const float* sinT   = (const float*)d_in[6];
    // d_in[7] = additive mask (causality applied analytically in-kernel)
    const int* save_slots  = (const int*)d_in[9];
    const int* fetch_slots = (const int*)d_in[10];
    const int* emb         = (const int*)d_in[11];

    const int T      = in_sizes[0] / (H_ * D_);
    const int B      = in_sizes[8];
    const int S      = T / B;
    const int NSLOTS = in_sizes[3] / (H_ * D_);

    // workspace layout: inv map | Kf (f16) | Vt (f16)
    char* ws = (char*)d_ws;
    int*      inv = (int*)ws;
    size_t off = ((size_t)NSLOTS * sizeof(int) + 255) & ~(size_t)255;
    _Float16* Kf = (_Float16*)(ws + off);
    off += (size_t)T * H_ * D_ * sizeof(_Float16);
    off = (off + 255) & ~(size_t)255;
    _Float16* Vt = (_Float16*)(ws + off);

    init_inv_kernel<<<(NSLOTS + 255) / 256, 256, 0, stream>>>(inv, NSLOTS);
    scatter_inv_kernel<<<(T + 255) / 256, 256, 0, stream>>>(save_slots, inv, T);

    dim3 ggrid((S + 31) / 32, H_, B);
    gather_kv_kernel<<<ggrid, 128, 0, stream>>>(K, V, Kcache, Vcache, cosT, sinT,
                                                fetch_slots, emb, inv, Kf, Vt, S);

    dim3 agrid((S + 63) / 64, H_, B);
    attn_kernel<<<agrid, 128, 0, stream>>>(Q, cosT, sinT, emb, Kf, Vt,
                                           (float*)d_out, S);
}
